// ProxyGML_4956392259677
// MI455X (gfx1250) — matrix operand order repaired and checked by reading, not compiled, not run on hardware
//
#include <hip/hip_runtime.h>
#include <hip/hip_bf16.h>
#include <math.h>

#define C_CLASSES 1024
#define ALLNUM    8192
#define DIM       512
#define TOPK      410
#define LAM       0.3f
#define BATCH     2048

typedef __bf16 bf16_t;
typedef __attribute__((ext_vector_type(8)))  __bf16 bf16x8;
typedef __attribute__((ext_vector_type(16))) __bf16 v16bf;
typedef __attribute__((ext_vector_type(8)))  float  v8f;

union V16U {
    struct { bf16x8 lo; bf16x8 hi; } p;
    v16bf v;
};

// ---- WMMA tile loaders (leading dim = DIM, row-major source) -------------
// A-matrix 16x32 bf16 layout (ISA 7.12.2): lanes 0-15: M=lane, K = k0+0..7 (VGPR0-3)
// and k0+16..23 (VGPR4-7); lanes 16-31: same M, K = k0+8..15 / k0+24..31.
__device__ __forceinline__ v16bf load_tile_A(const bf16_t* base, int row0, int k0, int lane) {
    int r  = row0 + (lane & 15);
    int kb = k0 + ((lane & 16) ? 8 : 0);
    const bf16_t* p = base + (size_t)r * DIM + kb;
    V16U u;
    u.p.lo = *(const bf16x8*)(p);
    u.p.hi = *(const bf16x8*)(p + 16);
    return u.v;
}
// B-matrix 32x16 bf16 layout: lane = N (mod 16); lanes 0-15 hold K=k0+0..7 / k0+16..23,
// lanes 16-31 hold K=k0+8..15 / k0+24..31 (same order as A). Loaded from B^T
// (N-major, K-contiguous) buffer.
__device__ __forceinline__ v16bf load_tile_B(const bf16_t* baseT, int col0, int k0, int lane) {
    int n  = col0 + (lane & 15);
    int kb = k0 + ((lane & 16) ? 8 : 0);
    const bf16_t* p = baseT + (size_t)n * DIM + kb;
    V16U u;
    u.p.lo = *(const bf16x8*)(p);
    u.p.hi = *(const bf16x8*)(p + 16);
    return u.v;
}

__device__ __forceinline__ v8f wmma_bf16(v16bf a, v16bf b, v8f c) {
    return __builtin_amdgcn_wmma_f32_16x16x32_bf16(false, a, false, b, (short)0, c,
                                                   false, false);
}

// ---- init & conversion ----------------------------------------------------
__global__ void k_init(float* acc) {
    if (threadIdx.x < 2) acc[threadIdx.x] = 0.0f;
}

__global__ void k_cvt(const float* __restrict__ x, bf16_t* __restrict__ y, int n) {
    int i = blockIdx.x * 256 + threadIdx.x;
    if (i < n) y[i] = (bf16_t)x[i];
}

// Column L2-normalize Proxies [DIM x ALLNUM], write transposed bf16 centers
// cT [ALLNUM x DIM] (row-major). Reads are fully coalesced across the block.
__global__ void k_norm(const float* __restrict__ P, bf16_t* __restrict__ cT) {
    int j = blockIdx.x * 256 + threadIdx.x;  // proxy column
    float s = 0.0f;
    for (int d = 0; d < DIM; ++d) {
        float v = P[(size_t)d * ALLNUM + j];
        s += v * v;
    }
    float inv = 1.0f / fmaxf(sqrtf(s), 1e-12f);
    for (int d = 0; d < DIM; ++d)
        cT[(size_t)j * DIM + d] = (bf16_t)(P[(size_t)d * ALLNUM + j] * inv);
}

// ---- similarity GEMM: sim[B x ALLNUM] = Xb[B x DIM] * centers --------------
// One wave computes a 16x64 output slab. Full-K A stripe hoisted into 128
// VGPRs; 4 independent accumulator chains kept live per k-step so B loads of
// one chain overlap the WMMAs of the others.
__global__ __launch_bounds__(256) void k_sim(const bf16_t* __restrict__ Xb,
                                             const bf16_t* __restrict__ cT,
                                             float* __restrict__ sim) {
    int lane = threadIdx.x & 31;
    int wave = threadIdx.x >> 5;
    int m0    = blockIdx.y * 16;
    int nbase = (blockIdx.x * 8 + wave) * 64;

    v16bf a_reg[DIM / 32];
#pragma unroll
    for (int ks = 0; ks < DIM / 32; ++ks)
        a_reg[ks] = load_tile_A(Xb, m0, ks * 32, lane);

    v8f acc[4];
#pragma unroll
    for (int q = 0; q < 4; ++q) acc[q] = (v8f){};

#pragma unroll
    for (int ks = 0; ks < DIM / 32; ++ks) {
        v16bf b0 = load_tile_B(cT, nbase +  0, ks * 32, lane);
        v16bf b1 = load_tile_B(cT, nbase + 16, ks * 32, lane);
        v16bf b2 = load_tile_B(cT, nbase + 32, ks * 32, lane);
        v16bf b3 = load_tile_B(cT, nbase + 48, ks * 32, lane);
        acc[0] = wmma_bf16(a_reg[ks], b0, acc[0]);
        acc[1] = wmma_bf16(a_reg[ks], b1, acc[1]);
        acc[2] = wmma_bf16(a_reg[ks], b2, acc[2]);
        acc[3] = wmma_bf16(a_reg[ks], b3, acc[3]);
    }
    // C/D layout: VGPR r, lane l -> row m0 + r + (l>=16 ? 8:0), col n0 + (l&15)
    int rbase = m0 + ((lane & 16) ? 8 : 0);
#pragma unroll
    for (int q = 0; q < 4; ++q) {
        int col = nbase + q * 16 + (lane & 15);
#pragma unroll
        for (int r = 0; r < 8; ++r)
            sim[(size_t)(rbase + r) * ALLNUM + col] = acc[q][r];
    }
}

// ---- per-row top-k + masked softmax CE ------------------------------------
__global__ __launch_bounds__(256) void k_classify(const float* __restrict__ sim,
                                                  const int* __restrict__ target,
                                                  float* __restrict__ accCls) {
    const int b = blockIdx.x;
    const int t = threadIdx.x;  // 256 threads
    __shared__ float srow[ALLNUM];
    __shared__ unsigned char sel[ALLNUM];
    __shared__ unsigned hist[256];
    __shared__ float red[256];
    __shared__ unsigned scanbuf[256];
    __shared__ unsigned sh_prefix;
    __shared__ int sh_kk;
    __shared__ float sh_et;

    const int tgt = target[b];
    for (int i = t; i < ALLNUM; i += 256) srow[i] = sim[(size_t)b * ALLNUM + i];
    if (t == 0) { sh_prefix = 0u; sh_kk = TOPK; sh_et = 0.0f; }
    __syncthreads();

    // exact radix-select of the TOPK-th largest sortable key (sim + 1000*pos)
    const unsigned himask[4] = {0u, 0xFF000000u, 0xFFFF0000u, 0xFFFFFF00u};
    for (int p = 0; p < 4; ++p) {
        int shift = 24 - 8 * p;
        hist[t] = 0u;
        __syncthreads();
        unsigned pref = sh_prefix;
        for (int i = t; i < ALLNUM; i += 256) {
            float x = srow[i] + (((i >> 3) == tgt) ? 1000.0f : 0.0f);
            unsigned u = __float_as_uint(x);
            unsigned key = (u & 0x80000000u) ? ~u : (u | 0x80000000u);
            if (((key ^ pref) & himask[p]) == 0u)
                atomicAdd(&hist[(key >> shift) & 255u], 1u);
        }
        __syncthreads();
        if (t == 0) {
            int kk = sh_kk; unsigned cum = 0; int bsel = 0;
            for (int bb = 255; bb >= 0; --bb) {
                unsigned c = hist[bb];
                if (cum + c >= (unsigned)kk) { bsel = bb; break; }
                cum += c;
            }
            sh_kk = kk - (int)cum;
            sh_prefix = pref | ((unsigned)bsel << shift);
        }
        __syncthreads();
    }
    const unsigned T = sh_prefix;
    const int ties_needed = sh_kk;

    // mark key > T; rank equal keys by index (top_k tie order) via prefix scan
    int cnt = 0;
    for (int i = t * 32; i < t * 32 + 32; ++i) {
        float x = srow[i] + (((i >> 3) == tgt) ? 1000.0f : 0.0f);
        unsigned u = __float_as_uint(x);
        unsigned key = (u & 0x80000000u) ? ~u : (u | 0x80000000u);
        sel[i] = (key > T) ? 1 : 0;
        if (key == T) ++cnt;
    }
    scanbuf[t] = (unsigned)cnt;
    __syncthreads();
    if (t == 0) {
        unsigned run = 0;
        for (int i = 0; i < 256; ++i) { unsigned c = scanbuf[i]; scanbuf[i] = run; run += c; }
    }
    __syncthreads();
    unsigned rank = scanbuf[t];
    for (int i = t * 32; i < t * 32 + 32; ++i) {
        float x = srow[i] + (((i >> 3) == tgt) ? 1000.0f : 0.0f);
        unsigned u = __float_as_uint(x);
        unsigned key = (u & 0x80000000u) ? ~u : (u | 0x80000000u);
        if (key == T) { if ((int)rank < ties_needed) sel[i] = 1; ++rank; }
    }
    __syncthreads();

    // logits[c] = sum of selected sims over the class's 8 proxies; masked softmax
    float lsum = 0.0f;
    for (int q = 0; q < 4; ++q) {
        int c = t + q * 256;
        float s = 0.0f;
        for (int n = 0; n < 8; ++n) { int j = c * 8 + n; if (sel[j]) s += srow[j]; }
        float e = (s != 0.0f) ? expf(s) : 0.0f;  // exp(logits) * (logits != 0)
        if (c == tgt) sh_et = e;
        lsum += e;
    }
    red[t] = lsum;
    __syncthreads();
    for (int s = 128; s > 0; s >>= 1) { if (t < s) red[t] += red[t + s]; __syncthreads(); }
    if (t == 0) {
        float predict = sh_et / (1e-8f + red[0]);
        atomicAdd(accCls, -logf(predict + 1e-20f));
    }
}

// ---- regularizer: streamed Gram rows + streaming softmax CE ----------------
// A stripe (rows i0..i0+15 of centers^T) hoisted into 128 VGPRs; each wave
// keeps 4 independent column-tile accumulator chains live so B loads overlap
// WMMA execution. Class logits are bounded (|v| <= ~8, cosine sums), so the
// exp-sum needs no running max: s += exp(v), logp = x_label - log(S).
__global__ __launch_bounds__(128) void k_reg(const bf16_t* __restrict__ cT,
                                             float* __restrict__ accReg) {
    const int t = threadIdx.x;  // 128 threads = 4 waves
    const int lane = t & 31, wave = t >> 5;
    const int i0 = blockIdx.x * 16;  // 16-row stripe of centers^T @ centers
    __shared__ float sbuf[16][8];
    __shared__ float xlab[16];

    v16bf a_reg[DIM / 32];
#pragma unroll
    for (int ks = 0; ks < DIM / 32; ++ks)
        a_reg[ks] = load_tile_A(cT, i0, ks * 32, lane);

    float s[8];
#pragma unroll
    for (int r = 0; r < 8; ++r) s[r] = 0.0f;

    // wave w handles jt = p*16 + w*4 + q (q = 0..3): disjoint, complete cover
    for (int p = 0; p < (ALLNUM / 16) / 16; ++p) {
        const int jtb = p * 16 + wave * 4;
        v8f acc[4];
#pragma unroll
        for (int q = 0; q < 4; ++q) acc[q] = (v8f){};
#pragma unroll
        for (int ks = 0; ks < DIM / 32; ++ks) {
            v16bf a  = a_reg[ks];
            v16bf b0 = load_tile_B(cT, (jtb + 0) * 16, ks * 32, lane);
            v16bf b1 = load_tile_B(cT, (jtb + 1) * 16, ks * 32, lane);
            v16bf b2 = load_tile_B(cT, (jtb + 2) * 16, ks * 32, lane);
            v16bf b3 = load_tile_B(cT, (jtb + 3) * 16, ks * 32, lane);
            acc[0] = wmma_bf16(a, b0, acc[0]);
            acc[1] = wmma_bf16(a, b1, acc[1]);
            acc[2] = wmma_bf16(a, b2, acc[2]);
            acc[3] = wmma_bf16(a, b3, acc[3]);
        }
        // Tile jt covers classes 2jt (cols 0-7) and 2jt+1 (cols 8-15).
        // Fold 8-lane column groups with xor shuffles -> per-(row,class) sums.
#pragma unroll
        for (int q = 0; q < 4; ++q) {
            const int jt = jtb + q;
            bool isLabelTile = (jt == blockIdx.x);  // labels of rows i0..i0+15
#pragma unroll
            for (int r = 0; r < 8; ++r) {
                float v = acc[q][r];
                v += __shfl_xor(v, 1);
                v += __shfl_xor(v, 2);
                v += __shfl_xor(v, 4);
                s[r] += expf(v);  // bounded logits: no running max needed
                if (isLabelTile) {
                    if (lane == 0)  xlab[r]     = v;  // row r,   class 2jt
                    if (lane == 24) xlab[8 + r] = v;  // row 8+r, class 2jt+1
                }
            }
        }
    }
    // merge 8 partial exp-sums per row: lanes {0,8,16,24} of each wave
    int row_base = (lane & 16) ? 8 : 0;
    int slot = wave * 2 + ((lane & 8) ? 1 : 0);
    if ((lane & 7) == 0) {
#pragma unroll
        for (int r = 0; r < 8; ++r) sbuf[row_base + r][slot] = s[r];
    }
    __syncthreads();
    if (t < 16) {
        float S = 0.0f;
        for (int q = 0; q < 8; ++q) S += sbuf[t][q];
        float logp = xlab[t] - logf(S);
        atomicAdd(accReg, -logp);
    }
}

__global__ void k_final(const float* __restrict__ acc, float* __restrict__ out) {
    float lc = acc[0] / (float)BATCH;
    float rg = acc[1] / (float)ALLNUM;
    out[0] = lc + LAM * rg;
    out[1] = lc;
}

// ---- host-side launch ------------------------------------------------------
extern "C" void kernel_launch(void* const* d_in, const int* in_sizes, int n_in,
                              void* d_out, int out_size, void* d_ws, size_t ws_size,
                              hipStream_t stream) {
    const float* x      = (const float*)d_in[0];  // [B, DIM]
    const int*   target = (const int*)d_in[1];    // [B]
    const float* P      = (const float*)d_in[2];  // [DIM, ALLNUM]
    // d_in[3] instance_label is repeat(arange(C), 8) -> label(j) = j >> 3, used implicitly

    const size_t SIM_BYTES = (size_t)BATCH * ALLNUM * sizeof(float);   // 64 MiB
    const size_t CT_BYTES  = (size_t)ALLNUM * DIM * sizeof(bf16_t);    //  8 MiB
    const size_t XB_BYTES  = (size_t)BATCH * DIM * sizeof(bf16_t);     //  2 MiB
    char* ws = (char*)d_ws;
    float*  sim = (float*)ws;
    bf16_t* cT  = (bf16_t*)(ws + SIM_BYTES);
    bf16_t* Xb  = (bf16_t*)(ws + SIM_BYTES + CT_BYTES);
    float*  acc = (float*)(ws + SIM_BYTES + CT_BYTES + XB_BYTES);      // [2]

    k_init<<<1, 32, 0, stream>>>(acc);
    k_cvt<<<(BATCH * DIM + 255) / 256, 256, 0, stream>>>(x, Xb, BATCH * DIM);
    k_norm<<<ALLNUM / 256, 256, 0, stream>>>(P, cT);
    k_sim<<<dim3(ALLNUM / 512, BATCH / 16), 256, 0, stream>>>(Xb, cT, sim);
    k_classify<<<BATCH, 256, 0, stream>>>(sim, target, acc + 0);
    k_reg<<<ALLNUM / 16, 128, 0, stream>>>(cT, acc + 1);
    k_final<<<1, 1, 0, stream>>>(acc, (float*)d_out);
}
